// MLPGibbsSampler_72086731096426
// MI455X (gfx1250) — matrix operand order repaired and checked by reading, not compiled, not run on hardware
//
#include <hip/hip_runtime.h>
#include <hip/hip_bf16.h>

// ---------------------------------------------------------------------------
// MLP Gibbs sampler for MI455X (gfx1250, wave32, WMMA).
//
// Per step (x32, stream-serialized):
//   1) h1 = relu(x  @ W1 + b1)   GEMM 2048x1024x1024, bf16 WMMA, bf16 out
//   2) h2 = relu(h1 @ W2 + b2)   GEMM 2048x1024x1024, bf16 WMMA, bf16 out
//   3) lg = h2 @ W3 + b3         GEMM 2048x1024x1024, bf16 WMMA, f32 out
//   4) per-row log-softmax, gather at tpos, total_log += ..., flip bit
//
// All weights (6 MB bf16) + activations (~16 MB) are L2-resident (192 MB),
// so fragments are loaded straight from global through WGP$/L2 with
// GLOBAL (not FLAT) loads: addresses are formed as GEPs on the kernel
// arguments each iteration so clang's address-space inference emits
// global_load_b128 / global_prefetch_b8 (FLAT would tie up DScnt + LDS path).
// Wave tile 32x64 -> 8 back-to-back v_wmma_f32_16x16x32_bf16 per K-step.
// ---------------------------------------------------------------------------

#define kB     2048   // batch
#define kD     1024   // discrete dim
#define kH     1024   // hidden (2*embed = 2*512)
#define kSteps 32

typedef __attribute__((ext_vector_type(16))) __bf16 v16bf;
typedef __attribute__((ext_vector_type(8)))  __bf16 v8bf;
typedef __attribute__((ext_vector_type(8)))  float  v8f;

// ---------------------------------------------------------------------------
// Weight convert + transpose: W [K x N] f32 (row-major) -> Wt [N x K] bf16.
// Transposed storage makes the WMMA B-fragment (one column per lane,
// 16 contiguous K values) a single contiguous 32-byte load.
// ---------------------------------------------------------------------------
__global__ __launch_bounds__(256) void wt_convert_kernel(
    const float* __restrict__ W, __bf16* __restrict__ Wt, int K, int N) {
  int idx = blockIdx.x * 256 + threadIdx.x;
  if (idx >= K * N) return;
  int k = idx / N;
  int n = idx - k * N;
  Wt[(size_t)n * K + k] = (__bf16)W[idx];
}

// ---------------------------------------------------------------------------
// Init: samples copies (f32 working copy == final_samples out region, plus
// very_init copy, plus bf16 activation copy), delta = float(target_pos),
// total_log = 0.
// ---------------------------------------------------------------------------
__global__ __launch_bounds__(256) void init_kernel(
    const float* __restrict__ init_samples,
    const int* __restrict__ target_pos,
    float* __restrict__ out_samples,
    float* __restrict__ out_delta,
    float* __restrict__ out_total_log,
    float* __restrict__ out_very_init,
    __bf16* __restrict__ xbf) {
  int idx = blockIdx.x * 256 + threadIdx.x;
  if (idx < kB * kD) {
    float v = init_samples[idx];
    out_samples[idx]   = v;
    out_very_init[idx] = v;
    xbf[idx]           = (__bf16)v;
  }
  if (idx < kSteps * kB) out_delta[idx] = (float)target_pos[idx];
  if (idx < kB)          out_total_log[idx] = 0.0f;
}

// ---------------------------------------------------------------------------
// bf16 WMMA GEMM:  C[M x N] = A[M x K] @ Bt^T  (+ bias, optional relu)
//   A  : bf16 row-major [M x K]
//   Bt : bf16 row-major [N x K]  (i.e. B transposed)
// Block: 256 threads = 8 waves (4 along M x 2 along N), C tile 128 x 128.
// Wave tile: 32(M) x 64(N) = 2x4 subtiles, 8 WMMAs / 12 b128 loads per step.
// ---------------------------------------------------------------------------
__global__ __launch_bounds__(256) void gemm_bf16_kernel(
    const __bf16* __restrict__ A,
    const __bf16* __restrict__ Bt,
    const float*  __restrict__ bias,
    __bf16* __restrict__ outBf,   // used when outIsBf16 != 0
    float*  __restrict__ outF,    // used when outIsBf16 == 0
    int M, int N, int K, int doRelu, int outIsBf16) {
  const int lane  = threadIdx.x & 31;
  const int wave  = threadIdx.x >> 5;
  const int waveM = wave & 3;                 // 4 waves along M
  const int waveN = wave >> 2;                // 2 waves along N
  const int m0    = blockIdx.y * 128 + waveM * 32;
  const int n0    = blockIdx.x * 128 + waveN * 64;
  const int half  = lane >> 4;                // 0: lanes 0-15, 1: lanes 16-31
  const int lid   = lane & 15;

  // Per-lane fragment element offsets (scalars after SROA; addresses are
  // rebuilt from A/Bt each iteration so loads stay in the GLOBAL aspace).
  size_t aoff[2];
#pragma unroll
  for (int mt = 0; mt < 2; ++mt)
    aoff[mt] = (size_t)(m0 + mt * 16 + lid) * K + half * 16;
  size_t boff[4];
#pragma unroll
  for (int nt = 0; nt < 4; ++nt)
    boff[nt] = (size_t)(n0 + nt * 16 + lid) * K + half * 16;

  v8f acc[2][4] = {};

  for (int k0 = 0; k0 < K; k0 += 32) {
    // A fragments (16x32 bf16): lane holds row (m%16), 16 contiguous K
    // values starting at k0 + half*16.
    v16bf afrag[2];
#pragma unroll
    for (int mt = 0; mt < 2; ++mt) {
      const __bf16* p = A + (aoff[mt] + k0);
      v8bf lo = *(const v8bf*)(p);
      v8bf hi = *(const v8bf*)(p + 8);
      afrag[mt] = __builtin_shufflevector(
          lo, hi, 0, 1, 2, 3, 4, 5, 6, 7, 8, 9, 10, 11, 12, 13, 14, 15);
      __builtin_prefetch(p + 32, 0, 1);  // unconditional, speculative
    }
    // B fragments (32x16 bf16 from transposed weights): lane holds column
    // (n%16), 16 contiguous K values starting at k0 + half*16.
    v16bf bfrag[4];
#pragma unroll
    for (int nt = 0; nt < 4; ++nt) {
      const __bf16* p = Bt + (boff[nt] + k0);
      bfrag[nt] = *(const v16bf*)(p);
      __builtin_prefetch(p + 32, 0, 1);
    }
    // 8 back-to-back matrix ops per K-step.
#pragma unroll
    for (int mt = 0; mt < 2; ++mt)
#pragma unroll
      for (int nt = 0; nt < 4; ++nt)
        acc[mt][nt] = __builtin_amdgcn_wmma_f32_16x16x32_bf16(
            false, afrag[mt], false, bfrag[nt], (short)0, acc[mt][nt],
            false, false);
  }

  // Epilogue: bias + relu, write bf16 or f32.
  // D layout: vgpr v, lanes 0-15 -> M = v, lanes 16-31 -> M = v + 8; N = lid.
#pragma unroll
  for (int nt = 0; nt < 4; ++nt) {
    const int   col = n0 + nt * 16 + lid;
    const float bv  = bias[col];
#pragma unroll
    for (int mt = 0; mt < 2; ++mt) {
      v8f d = acc[mt][nt];
#pragma unroll
      for (int v = 0; v < 8; ++v) {
        float val = d[v] + bv;
        if (doRelu) val = fmaxf(val, 0.0f);
        const size_t row = (size_t)(m0 + mt * 16 + v + half * 8);
        if (outIsBf16) outBf[row * N + col] = (__bf16)val;
        else           outF [row * N + col] = val;
      }
    }
  }
}

// ---------------------------------------------------------------------------
// Per-row log-softmax + gather + accumulate + bit flip. One block per row.
// ---------------------------------------------------------------------------
__global__ __launch_bounds__(256) void softmax_gather_flip_kernel(
    const float* __restrict__ logits,      // [kB x kD]
    const int*   __restrict__ target_pos,  // [kSteps x kB]
    int step,
    float*  __restrict__ samples,          // [kB x kD] (== final_samples out)
    __bf16* __restrict__ xbf,              // [kB x kD] bf16 mirror
    float*  __restrict__ total_log) {      // [kB]
  __shared__ float red[256];
  const int row = blockIdx.x;
  const float* lr = logits + (size_t)row * kD;

  float m = -3.4e38f;
  for (int c = threadIdx.x; c < kD; c += 256) m = fmaxf(m, lr[c]);
  red[threadIdx.x] = m;
  __syncthreads();
  for (int s = 128; s > 0; s >>= 1) {
    if (threadIdx.x < s)
      red[threadIdx.x] = fmaxf(red[threadIdx.x], red[threadIdx.x + s]);
    __syncthreads();
  }
  m = red[0];
  __syncthreads();

  float sum = 0.0f;
  for (int c = threadIdx.x; c < kD; c += 256) sum += __expf(lr[c] - m);
  red[threadIdx.x] = sum;
  __syncthreads();
  for (int s = 128; s > 0; s >>= 1) {
    if (threadIdx.x < s) red[threadIdx.x] += red[threadIdx.x + s];
    __syncthreads();
  }

  if (threadIdx.x == 0) {
    const float logZ = m + __logf(red[0]);
    const int   tp   = target_pos[step * kB + row];
    total_log[row] += lr[tp] - logZ;
    const size_t off = (size_t)row * kD + tp;
    const float nb = 1.0f - samples[off];
    samples[off] = nb;
    xbf[off]     = (__bf16)nb;
  }
}

// ---------------------------------------------------------------------------
// Host-side orchestration (graph-capture safe: launches only).
// ---------------------------------------------------------------------------
extern "C" void kernel_launch(void* const* d_in, const int* in_sizes, int n_in,
                              void* d_out, int out_size, void* d_ws,
                              size_t ws_size, hipStream_t stream) {
  (void)in_sizes; (void)n_in; (void)out_size; (void)ws_size;

  const float* init_samples = (const float*)d_in[0];
  const int*   target_pos   = (const int*)d_in[1];
  const float* W1 = (const float*)d_in[2];
  const float* b1 = (const float*)d_in[3];
  const float* W2 = (const float*)d_in[4];
  const float* b2 = (const float*)d_in[5];
  const float* W3 = (const float*)d_in[6];
  const float* b3 = (const float*)d_in[7];

  // Output layout: final_samples | delta | total_log | very_init_samples
  float* out_samples  = (float*)d_out;
  float* out_delta    = out_samples + (size_t)kB * kD;
  float* out_total    = out_delta + (size_t)kSteps * kB;
  float* out_veryinit = out_total + kB;

  // Workspace carve-out.
  char*  ws = (char*)d_ws;
  size_t o  = 0;
  auto carve = [&](size_t bytes) -> void* {
    void* p = ws + o;
    o += (bytes + 255) & ~(size_t)255;
    return p;
  };
  __bf16* W1t    = (__bf16*)carve((size_t)kH * kD * 2);  // [kH x kD]
  __bf16* W2t    = (__bf16*)carve((size_t)kH * kH * 2);  // [kH x kH]
  __bf16* W3t    = (__bf16*)carve((size_t)kD * kH * 2);  // [kD x kH]
  __bf16* xbf    = (__bf16*)carve((size_t)kB * kD * 2);  // [kB x kD]
  __bf16* h1     = (__bf16*)carve((size_t)kB * kH * 2);  // [kB x kH]
  __bf16* h2     = (__bf16*)carve((size_t)kB * kH * 2);  // [kB x kH]
  float*  logits = (float*) carve((size_t)kB * kD * 4);  // [kB x kD]

  // --- Prep: weight convert/transpose + output/activation init.
  {
    int n1 = kD * kH;
    wt_convert_kernel<<<(n1 + 255) / 256, 256, 0, stream>>>(W1, W1t, kD, kH);
    int n2 = kH * kH;
    wt_convert_kernel<<<(n2 + 255) / 256, 256, 0, stream>>>(W2, W2t, kH, kH);
    int n3 = kH * kD;
    wt_convert_kernel<<<(n3 + 255) / 256, 256, 0, stream>>>(W3, W3t, kH, kD);
    int ni = kB * kD;
    init_kernel<<<(ni + 255) / 256, 256, 0, stream>>>(
        init_samples, target_pos, out_samples, out_delta, out_total,
        out_veryinit, xbf);
  }

  // --- 32 Gibbs steps, stream-serialized.
  const dim3 gBlk(256);
  const dim3 gGrid(kH / 128, kB / 128);  // (N/128, M/128); N==kH==kD==1024
  for (int step = 0; step < kSteps; ++step) {
    // layer 1: h1 = relu(x @ W1 + b1)
    gemm_bf16_kernel<<<gGrid, gBlk, 0, stream>>>(
        xbf, W1t, b1, h1, nullptr, kB, kH, kD, /*relu=*/1, /*bf16=*/1);
    // layer 2: h2 = relu(h1 @ W2 + b2)
    gemm_bf16_kernel<<<gGrid, gBlk, 0, stream>>>(
        h1, W2t, b2, h2, nullptr, kB, kH, kH, /*relu=*/1, /*bf16=*/1);
    // layer 3: logits = h2 @ W3 + b3
    gemm_bf16_kernel<<<gGrid, gBlk, 0, stream>>>(
        h2, W3t, b3, nullptr, logits, kB, kD, kH, /*relu=*/0, /*bf16=*/0);
    // log-softmax + gather + accumulate + flip
    softmax_gather_flip_kernel<<<kB, gBlk, 0, stream>>>(
        logits, target_pos, step, out_samples, xbf, out_total);
  }
}
